// Inference_3100966388277
// MI455X (gfx1250) — compile-verified, matrix operand order checked by reading, not run on hardware
//
#include <hip/hip_runtime.h>
#include <hip/hip_bf16.h>

// ---------------------------------------------------------------------------
// RNN-T greedy decode for MI455X (gfx1250, wave32, WMMA).
//   B=32, T=512, D_ENC=512, H=E=J=320, V=1024, STEPS=1024
// Strategy:
//   * Pre-swizzle all weight matrices to bf16 in the exact WMMA B-operand
//     per-lane layout -> one 32B load per lane per 32x16 B tile.
//   * enc_proj = enc^T @ We as a parallel WMMA GEMM into workspace
//     (B tiles double-buffered across k-steps).
//   * One persistent 512-thread (16-wave) workgroup runs the 1024-step scan:
//     LSTM / joint / logits via v_wmma_f32_16x16x32_bf16, h/c state in
//     registers, activations staged in LDS, Wp staged in LDS once (reused
//     1024x; 320KB WGP LDS), argmax via shuffle + ds_max_u64.
//   * Joint/logits pair both M tiles per N tile: one shared B tile feeds two
//     independent WMMA chains (2x ILP, half the B traffic).
// ---------------------------------------------------------------------------

typedef __attribute__((ext_vector_type(16))) __bf16 v16bf;
typedef __attribute__((ext_vector_type(8)))  __bf16 v8bf;
typedef __attribute__((ext_vector_type(8)))  float  v8f;

#define BB      32
#define TT      512
#define DENC    512
#define HH      320
#define VV      1024
#define STEPS   1024
#define FOURH   1280

__device__ __forceinline__ v8f wmma_bf16(v16bf a, v16bf b, v8f c) {
    return __builtin_amdgcn_wmma_f32_16x16x32_bf16(
        false, a, false, b, (short)0, c, false, false);
}

// A operand (16x32 bf16) from a row-major LDS buffer [rows][ld].
// chunk0 = k in [ks*32 + kg*8, +8), chunk1 = k in [ks*32 + 16 + kg*8, +8).
__device__ __forceinline__ v16bf lds_load_A(const __bf16* buf, int ld,
                                            int mtile, int kstep, int lane) {
    int m  = mtile * 16 + (lane & 15);
    int kg = (lane >> 4) & 1;
    const __bf16* p = buf + m * ld + kstep * 32 + kg * 8;
    v8bf c0 = *(const v8bf*)(p);
    v8bf c1 = *(const v8bf*)(p + 16);
    v16bf a;
#pragma unroll
    for (int i = 0; i < 8; ++i) { a[i] = c0[i]; a[i + 8] = c1[i]; }
    return a;
}

// B operand (32x16 bf16) from a pre-swizzled weight buffer (global):
// layout [ntile][kstep][lane][16 elems] -> one 32B contiguous load per lane.
__device__ __forceinline__ v16bf g_load_B(const __bf16* wsw, int ksteps,
                                          int ntile, int kstep, int lane) {
    const __bf16* p = wsw + ((((long)ntile * ksteps + kstep) * 32) + lane) * 16;
    return *(const v16bf*)p;
}

// Same layout, but from an LDS copy of the weights.
__device__ __forceinline__ v16bf lds_load_B(const __bf16* wsw, int ksteps,
                                            int ntile, int kstep, int lane) {
    const __bf16* p = wsw + (((ntile * ksteps + kstep) * 32) + lane) * 16;
    v8bf c0 = *(const v8bf*)(p);
    v8bf c1 = *(const v8bf*)(p + 8);
    v16bf b;
#pragma unroll
    for (int i = 0; i < 8; ++i) { b[i] = c0[i]; b[i + 8] = c1[i]; }
    return b;
}

__device__ __forceinline__ float sigmoidf(float x) {
    return 1.0f / (1.0f + __expf(-x));
}

// ---------------------------------------------------------------------------
// Swizzle f32 weight (K x N, row-major) into bf16 WMMA-B layout.
// ---------------------------------------------------------------------------
__global__ void swizzle_w_kernel(const float* __restrict__ src,
                                 __bf16* __restrict__ dst, int K, int N) {
    int total  = K * N;
    int ksteps = K / 32;
    for (int tid = blockIdx.x * blockDim.x + threadIdx.x; tid < total;
         tid += gridDim.x * blockDim.x) {
        int e     = tid & 15;
        int lane  = (tid >> 4) & 31;
        int kstep = (tid >> 9) % ksteps;
        int ntile = tid / (512 * ksteps);
        int v = e >> 1, half = e & 1;
        int kg = lane >> 4, nl = lane & 15;
        int kl = (v >> 2) * 16 + kg * 8 + (v & 3) * 2 + half;
        int k  = kstep * 32 + kl;
        int n  = ntile * 16 + nl;
        dst[tid] = (__bf16)src[(long)k * N + n];
    }
}

__global__ void to_bf16_kernel(const float* __restrict__ src,
                               __bf16* __restrict__ dst, int n) {
    for (int i = blockIdx.x * blockDim.x + threadIdx.x; i < n;
         i += gridDim.x * blockDim.x)
        dst[i] = (__bf16)src[i];
}

// ---------------------------------------------------------------------------
// enc_proj[b*T+t, :] = enc[b,:,t] @ We   (M = B*T = 16384, N = 320, K = 512)
// One block per 16-row M tile; 8 waves cover 20 N tiles; B double-buffered.
// ---------------------------------------------------------------------------
__global__ void __launch_bounds__(256)
encproj_kernel(const float* __restrict__ enc, const __bf16* __restrict__ We_sw,
               float* __restrict__ out) {
    __shared__ __align__(16) __bf16 abuf[16 * DENC];
    int mt  = blockIdx.x;          // 0..1023
    int tid = threadIdx.x;
    for (int i = tid; i < 16 * DENC; i += 256) {
        int m = i >> 9, d = i & 511;
        int gr = mt * 16 + m;
        int b = gr >> 9, t = gr & 511;          // gr = b*T + t
        abuf[i] = (__bf16)enc[((long)b * DENC + d) * TT + t];
    }
    __syncthreads();
    int wave = tid >> 5, lane = tid & 31;
    int kg = lane >> 4, nl = lane & 15;
    for (int nt = wave; nt < 20; nt += 8) {
        v8f acc = {};
        v16bf bcur = g_load_B(We_sw, 16, nt, 0, lane);
#pragma unroll
        for (int ks = 0; ks < 16; ++ks) {
            v16bf bnext = (ks < 15) ? g_load_B(We_sw, 16, nt, ks + 1, lane) : bcur;
            acc = wmma_bf16(lds_load_A(abuf, DENC, 0, ks, lane), bcur, acc);
            bcur = bnext;
        }
#pragma unroll
        for (int r = 0; r < 8; ++r) {
            int gr = mt * 16 + r + kg * 8;
            out[(long)gr * HH + nt * 16 + nl] = acc[r];
        }
    }
}

// ---------------------------------------------------------------------------
// Persistent sequential decode: one workgroup, 512 threads = 16 waves.
// ---------------------------------------------------------------------------
__global__ void __launch_bounds__(512)
decode_kernel(const float* __restrict__ enc_proj,
              const __bf16* __restrict__ embed_bf,
              const __bf16* __restrict__ Wx_sw,
              const __bf16* __restrict__ Wh_sw,
              const __bf16* __restrict__ Wp_sw,
              const __bf16* __restrict__ Wo_sw,
              const float* __restrict__ b_lstm,
              const float* __restrict__ b_joint,
              const float* __restrict__ bo,
              const int* __restrict__ enc_lens,
              int* __restrict__ labels_out,
              float* __restrict__ logits_out,
              float* __restrict__ h_out,
              float* __restrict__ c_out) {
    __shared__ __align__(16) __bf16 xbuf[BB * HH];     // embedded input (bf16)
    __shared__ __align__(16) __bf16 hbuf[BB * HH];     // h (pre/post mask)
    __shared__ __align__(16) __bf16 jbuf[BB * HH];     // joint activation
    __shared__ __align__(16) __bf16 wpbuf[HH * HH];    // Wp staged in LDS (200KB)
    __shared__ unsigned long long amax[BB];
    __shared__ int s_last[BB], s_time[BB], s_syms[BB], s_halt[BB], s_tidx[BB];

    int tid  = threadIdx.x;
    int wave = tid >> 5, lane = tid & 31;
    int kg = lane >> 4, nl = lane & 15;

    // Persistent per-wave LSTM state tiles (<=3 hidden jobs per wave).
    v8f hreg[3] = {}, creg[3] = {};
    v8f h2t[3], c2t[3];

    // One-time staging: zero h/x, copy Wp (swizzled) to LDS, init control.
    for (int i = tid; i < BB * HH; i += 512) {
        hbuf[i] = (__bf16)0.0f;
        xbuf[i] = (__bf16)0.0f;     // last == BLANK at step 0 -> x = 0
    }
    {
        uint4* d = (uint4*)wpbuf;
        const uint4* sgl = (const uint4*)Wp_sw;
        for (int i = tid; i < (HH * HH) / 8; i += 512) d[i] = sgl[i];
    }
    if (tid < BB) {
        s_last[tid] = 0; s_time[tid] = 0; s_syms[tid] = 0;
        s_halt[tid] = 0; s_tidx[tid] = 0;
        amax[tid] = 0ULL;
    }
    __syncthreads();

#pragma unroll 1
    for (int s = 0; s < STEPS; ++s) {
        // ---- Phase B: LSTM gates. 40 hidden jobs (2 Mtiles x 20 Ntiles) ----
#pragma unroll
        for (int jix = 0; jix < 3; ++jix) {
            int jj = wave + jix * 16;
            if (jj < 40) {
                int ntile = jj % 20, mtile = jj / 20;
                v8f ai = {}, af = {}, ag = {}, ao = {};
#pragma unroll
                for (int ks = 0; ks < 10; ++ks) {
                    v16bf ax = lds_load_A(xbuf, HH, mtile, ks, lane);
                    v16bf ah = lds_load_A(hbuf, HH, mtile, ks, lane);
                    ai = wmma_bf16(ax, g_load_B(Wx_sw, 10,  0 + ntile, ks, lane), ai);
                    ai = wmma_bf16(ah, g_load_B(Wh_sw, 10,  0 + ntile, ks, lane), ai);
                    af = wmma_bf16(ax, g_load_B(Wx_sw, 10, 20 + ntile, ks, lane), af);
                    af = wmma_bf16(ah, g_load_B(Wh_sw, 10, 20 + ntile, ks, lane), af);
                    ag = wmma_bf16(ax, g_load_B(Wx_sw, 10, 40 + ntile, ks, lane), ag);
                    ag = wmma_bf16(ah, g_load_B(Wh_sw, 10, 40 + ntile, ks, lane), ag);
                    ao = wmma_bf16(ax, g_load_B(Wx_sw, 10, 60 + ntile, ks, lane), ao);
                    ao = wmma_bf16(ah, g_load_B(Wh_sw, 10, 60 + ntile, ks, lane), ao);
                }
                int col = ntile * 16 + nl;
                float bi = b_lstm[0 * HH + col], bf_ = b_lstm[1 * HH + col];
                float bg = b_lstm[2 * HH + col], bo_ = b_lstm[3 * HH + col];
#pragma unroll
                for (int r = 0; r < 8; ++r) {
                    float iv = sigmoidf(ai[r] + bi);
                    float fv = sigmoidf(af[r] + bf_);
                    float gv = tanhf(ag[r] + bg);
                    float ov = sigmoidf(ao[r] + bo_);
                    float c2 = fv * creg[jix][r] + iv * gv;
                    float h2 = ov * tanhf(c2);
                    c2t[jix][r] = c2;
                    h2t[jix][r] = h2;
                    int m = mtile * 16 + r + kg * 8;
                    hbuf[m * HH + col] = (__bf16)h2;   // h2 (unmasked) for joint
                }
            }
        }
        __syncthreads();

        // ---- Phase C: joint = tanh(e + h2 @ Wp + b_joint) ----
        // Wp is LDS-resident; both M tiles share each B tile (2 chains).
#pragma unroll 1
        for (int nt = wave; nt < 20; nt += 16) {    // waves 0-3 run twice
            v8f acc0 = {}, acc1 = {};
#pragma unroll
            for (int ks = 0; ks < 10; ++ks) {
                v16bf bm = lds_load_B(wpbuf, 10, nt, ks, lane);
                acc0 = wmma_bf16(lds_load_A(hbuf, HH, 0, ks, lane), bm, acc0);
                acc1 = wmma_bf16(lds_load_A(hbuf, HH, 1, ks, lane), bm, acc1);
            }
            int col = nt * 16 + nl;
            float bj = b_joint[col];
#pragma unroll
            for (int r = 0; r < 8; ++r) {
                int m0 = r + kg * 8;                  // batch rows 0..15
                int m1 = 16 + r + kg * 8;             // batch rows 16..31
                float e0 = enc_proj[((long)m0 * TT + s_tidx[m0]) * HH + col];
                float e1 = enc_proj[((long)m1 * TT + s_tidx[m1]) * HH + col];
                jbuf[m0 * HH + col] = (__bf16)tanhf(acc0[r] + e0 + bj);
                jbuf[m1 * HH + col] = (__bf16)tanhf(acc1[r] + e1 + bj);
            }
        }
        __syncthreads();

        // ---- Phase D: logits = jbuf @ Wo + bo, write out, argmax ----
        // 64 N-tiles / 16 waves; both M tiles share B; B double-buffered.
#pragma unroll 1
        for (int nt = wave; nt < 64; nt += 16) {
            v8f acc0 = {}, acc1 = {};
            v16bf bcur = g_load_B(Wo_sw, 10, nt, 0, lane);
#pragma unroll
            for (int ks = 0; ks < 10; ++ks) {
                v16bf bnext = (ks < 9) ? g_load_B(Wo_sw, 10, nt, ks + 1, lane) : bcur;
                acc0 = wmma_bf16(lds_load_A(jbuf, HH, 0, ks, lane), bcur, acc0);
                acc1 = wmma_bf16(lds_load_A(jbuf, HH, 1, ks, lane), bcur, acc1);
                bcur = bnext;
            }
            int col = nt * 16 + nl;
            float bb = bo[col];
#pragma unroll
            for (int mtile = 0; mtile < 2; ++mtile) {
                v8f acc = mtile ? acc1 : acc0;
#pragma unroll
                for (int r = 0; r < 8; ++r) {
                    float lv = acc[r] + bb;
                    int m = mtile * 16 + r + kg * 8;          // batch index
                    logits_out[((long)m * STEPS + s) * VV + col] = lv;
                    // shuffle-reduce argmax across the 16 lanes of this row
                    float best = lv; int bcol = col;
#pragma unroll
                    for (int off = 1; off < 16; off <<= 1) {
                        float ov = __shfl_xor(best, off, 32);
                        int   oc = __shfl_xor(bcol, off, 32);
                        if (ov > best || (ov == best && oc < bcol)) {
                            best = ov; bcol = oc;
                        }
                    }
                    if (nl == 0) {
                        unsigned int kb = __float_as_uint(best);
                        kb = (kb & 0x80000000u) ? ~kb : (kb | 0x80000000u);
                        unsigned long long packed =
                            ((unsigned long long)kb << 32) |
                            (unsigned)(VV - 1 - bcol);
                        atomicMax(&amax[m], packed);   // ds_max_u64
                    }
                }
            }
        }
        __syncthreads();

        // ---- Phase E: per-batch control update ----
        if (tid < BB) {
            int b = tid;
            int sym    = (VV - 1) - (int)(amax[b] & 0xFFFFFFFFull);
            int active = s_time[b] < enc_lens[b];
            int halt   = (sym == 0) || (!active);
            int adv    = halt || (s_syms[b] + 1 >= 2);
            labels_out[(long)b * STEPS + s] = active ? sym : 0;
            s_time[b] += adv ? 1 : 0;
            s_syms[b]  = adv ? 0 : s_syms[b] + 1;
            s_last[b]  = halt ? s_last[b] : sym;
            s_halt[b]  = halt;
            int ti = s_time[b];
            s_tidx[b] = ti < (TT - 1) ? ti : (TT - 1);
            amax[b] = 0ULL;                       // reset for next step
        }
        __syncthreads();

        // ---- Phase F: commit masked state, rewrite hbuf, build next x ----
#pragma unroll
        for (int jix = 0; jix < 3; ++jix) {
            int jj = wave + jix * 16;
            if (jj < 40) {
                int ntile = jj % 20, mtile = jj / 20;
                int col = ntile * 16 + nl;
#pragma unroll
                for (int r = 0; r < 8; ++r) {
                    int m = mtile * 16 + r + kg * 8;
                    float nh = s_halt[m] ? hreg[jix][r] : h2t[jix][r];
                    float nc = s_halt[m] ? creg[jix][r] : c2t[jix][r];
                    hreg[jix][r] = nh;
                    creg[jix][r] = nc;
                    hbuf[m * HH + col] = (__bf16)nh;
                }
            }
        }
        for (int i = tid; i < BB * HH; i += 512) {    // x for next step
            int b = i / HH, col = i % HH;
            int lst = s_last[b];
            xbuf[i] = (lst == 0) ? (__bf16)0.0f : embed_bf[(long)lst * HH + col];
        }
        __syncthreads();
    }

    // ---- Final h, c outputs ----
#pragma unroll
    for (int jix = 0; jix < 3; ++jix) {
        int jj = wave + jix * 16;
        if (jj < 40) {
            int ntile = jj % 20, mtile = jj / 20;
            int col = ntile * 16 + nl;
#pragma unroll
            for (int r = 0; r < 8; ++r) {
                int m = mtile * 16 + r + kg * 8;
                h_out[(long)m * HH + col] = hreg[jix][r];
                c_out[(long)m * HH + col] = creg[jix][r];
            }
        }
    }
}

// ---------------------------------------------------------------------------
extern "C" void kernel_launch(void* const* d_in, const int* in_sizes, int n_in,
                              void* d_out, int out_size, void* d_ws, size_t ws_size,
                              hipStream_t stream) {
    const float* enc      = (const float*)d_in[0];   // (B, D_ENC, T)
    const int*   enc_lens = (const int*)  d_in[1];   // (B,)
    const float* embed    = (const float*)d_in[2];   // (V, E)
    const float* Wx       = (const float*)d_in[3];   // (E, 4H)
    const float* Wh       = (const float*)d_in[4];   // (H, 4H)
    const float* b_lstm   = (const float*)d_in[5];   // (4H,)
    const float* We       = (const float*)d_in[6];   // (D_ENC, J)
    const float* Wp       = (const float*)d_in[7];   // (H, J)
    const float* b_joint  = (const float*)d_in[8];   // (J,)
    const float* Wo       = (const float*)d_in[9];   // (J, V)
    const float* bo       = (const float*)d_in[10];  // (V,)

    // Workspace layout
    char* ws = (char*)d_ws;
    float*  enc_proj = (float*)ws;                       size_t off = (size_t)BB * TT * HH * 4;
    __bf16* Wx_sw    = (__bf16*)(ws + off);              off += (size_t)HH * FOURH * 2;
    __bf16* Wh_sw    = (__bf16*)(ws + off);              off += (size_t)HH * FOURH * 2;
    __bf16* Wp_sw    = (__bf16*)(ws + off);              off += (size_t)HH * HH * 2;
    __bf16* Wo_sw    = (__bf16*)(ws + off);              off += (size_t)HH * VV * 2;
    __bf16* We_sw    = (__bf16*)(ws + off);              off += (size_t)DENC * HH * 2;
    __bf16* embed_bf = (__bf16*)(ws + off);              off += (size_t)VV * HH * 2;

    // Output layout: labels (int32 bits) | logits | h | c
    int*   labels = (int*)d_out;
    float* logits = (float*)d_out + (size_t)BB * STEPS;
    float* hout   = logits + (size_t)BB * STEPS * VV;
    float* cout   = hout + (size_t)BB * HH;

    swizzle_w_kernel<<<400, 256, 0, stream>>>(Wx, Wx_sw, HH, FOURH);
    swizzle_w_kernel<<<400, 256, 0, stream>>>(Wh, Wh_sw, HH, FOURH);
    swizzle_w_kernel<<<100, 256, 0, stream>>>(Wp, Wp_sw, HH, HH);
    swizzle_w_kernel<<<320, 256, 0, stream>>>(Wo, Wo_sw, HH, VV);
    swizzle_w_kernel<<<160, 256, 0, stream>>>(We, We_sw, DENC, HH);
    to_bf16_kernel <<<320, 256, 0, stream>>>(embed, embed_bf, VV * HH);

    encproj_kernel<<<(BB * TT) / 16, 256, 0, stream>>>(enc, We_sw, enc_proj);

    decode_kernel<<<1, 512, 0, stream>>>(enc_proj, embed_bf, Wx_sw, Wh_sw,
                                         Wp_sw, Wo_sw, b_lstm, b_joint, bo,
                                         enc_lens, labels, logits, hout, cout);
}
